// _PositionAttentionModule_52055003627961
// MI455X (gfx1250) — matrix operand order repaired and checked by reading, not compiled, not run on hardware
//
#include <hip/hip_runtime.h>

typedef __attribute__((ext_vector_type(16))) _Float16 v16h;
typedef __attribute__((ext_vector_type(8)))  _Float16 v8h;
typedef __attribute__((ext_vector_type(8)))  float    v8f;
typedef unsigned int u32x4 __attribute__((ext_vector_type(4)));
typedef int          i32x8 __attribute__((ext_vector_type(8)));
typedef int          i32x4 __attribute__((ext_vector_type(4)));

#define B_  4
#define C_  256
#define CQ  32
#define N_  4096   // H*W = 64*64
#define DROW 40    // LDS row stride (halfs) for feat_d slab: 32 data + 8 pad

#if __has_builtin(__builtin_amdgcn_tensor_load_to_lds)
#define USE_TDM 1
#else
#define USE_TDM 0
#endif

// ---------------------------------------------------------------------------
// Projection: Y(f16) = oscale * (W[Co x 256] @ X[b][256][N] + bias)
// transposed=1 -> Y stored [B][N][Co]; transposed=0 -> Y stored [B][Co][N]
// ---------------------------------------------------------------------------
__global__ __launch_bounds__(256) void proj_kernel(
    const float* __restrict__ X, const float* __restrict__ Wm,
    const float* __restrict__ bias, _Float16* __restrict__ Y,
    int Co, int transposed, float oscale)
{
    __shared__ float xs[C_ * 64];              // 64 KB tile of X
    const int ntiles = N_ / 64;
    const int b  = blockIdx.x / ntiles;
    const int n0 = (blockIdx.x % ntiles) * 64;
    const int t  = threadIdx.x;

    const float* Xb = X + (size_t)b * C_ * N_;
    for (int idx = t; idx < C_ * 64; idx += 256) {
        int c = idx >> 6, j = idx & 63;
        xs[idx] = Xb[(size_t)c * N_ + n0 + j];
    }
    __syncthreads();

    const int j  = t & 63;
    const int og = t >> 6;                      // wave-uniform output group
    for (int o = og; o < Co; o += 4) {
        float acc = bias[o];
        const float* wrow = Wm + (size_t)o * C_;   // scalar (wave-uniform) loads
        #pragma unroll 8
        for (int c = 0; c < C_; ++c)
            acc = fmaf(wrow[c], xs[c * 64 + j], acc);
        _Float16 h = (_Float16)(acc * oscale);
        if (transposed) Y[(size_t)b * N_ * Co + (size_t)(n0 + j) * Co + o] = h;
        else            Y[(size_t)b * Co * N_ + (size_t)o  * N_ + n0 + j] = h;
    }
}

// ---------------------------------------------------------------------------
// TDM: DMA a [256 ch x 32 key] f16 slab of feat_d (row stride N_) into LDS,
// padded +4 dwords per 16-dword row (LDS row stride = DROW halfs).
// D# per CDNA5 ISA ch.8: group0 {count, lds_addr, global_addr, type=2},
// group1 {data_size=2B, pad_enable, pad_interval=16dw, pad_amount=4dw,
//         tensor_dims=2^30 (no OOB), tile 32x256, dim0 stride 4096}.
// ---------------------------------------------------------------------------
#if USE_TDM
__device__ __forceinline__ void tdm_load_slab(const _Float16* gsrc,
                                              _Float16* ldst)
{
    unsigned lds_addr      = (unsigned)(size_t)(void*)ldst;
    unsigned long long ga  = (unsigned long long)(size_t)gsrc;
    u32x4 g0;
    g0[0] = 1u;                                   // count=1, user descriptor
    g0[1] = lds_addr;                             // LDS byte address
    g0[2] = (unsigned)ga;                         // global_addr[31:0]
    g0[3] = (unsigned)((ga >> 32) & 0x1FFFFFFu)   // global_addr[56:32]
          | (2u << 30);                           // type = 2 ("image")
    i32x8 g1;
    g1[0] = (int)((1u << 16)      // data_size = 1 -> 2 bytes
                | (1u << 20)      // pad_enable
                | (3u << 22)      // pad_interval code 3 -> every 16 dwords
                | (3u << 25));    // pad_amount  code 3 -> 4 dwords
    g1[1] = 0;                    // no atomic barrier
    g1[2] = 0x00004000;           // tensor_dim0 = 2^30 (bits 79:48)
    g1[3] = 0x00204000;           // tensor_dim1 = 2^30 | tile_dim0 = 32
    g1[4] = 256;                  // tile_dim1 = 256 (tile_dim2 = 0)
    g1[5] = 4096;                 // tensor_dim0_stride = N_
    g1[6] = 0;                    // stride[47:32] = 0
    g1[7] = 0;                    // tensor_dim1_stride unused (2D tile)
    i32x4 gz = {0, 0, 0, 0};
#if __clang_major__ >= 23
    i32x8 gz8 = {0, 0, 0, 0, 0, 0, 0, 0};
    __builtin_amdgcn_tensor_load_to_lds(g0, g1, gz, gz, gz8, 0);
#else
    __builtin_amdgcn_tensor_load_to_lds(g0, g1, gz, gz, 0);
#endif
}
#endif

// ---------------------------------------------------------------------------
// Fused attention + PV + residual. One wave32 per 16-query tile, 4 waves/block.
// Softmax via shift-invariance (bounded logits): P = exp2(S2) raw, l = P@ones
// via WMMA; out = alpha * (P@D)/l + x. feat_d key-slabs are TDM-DMA'd into
// LDS once per block (double-buffered), consumed by all 4 waves.
//   featb  : [B][N][32]  f16, pre-scaled by log2(e)  (A matrix rows)
//   featcT : [B][N][32]  f16  (keys as rows -> B columns contiguous)
//   featd  : [B][256][N] f16  (channel rows -> PV B columns contiguous)
// ---------------------------------------------------------------------------
__global__ __launch_bounds__(128) void pam_attn_kernel(
    const _Float16* __restrict__ featb,
    const _Float16* __restrict__ featcT,
    const _Float16* __restrict__ featd,
    const float*    __restrict__ xin,
    const float*    __restrict__ alpha_p,
    float*          __restrict__ out)
{
    __shared__ __align__(16) _Float16 plds[4][16 * 32];   // P tile per wave
    __shared__ __align__(16) _Float16 dbuf[2][C_ * DROW]; // feat_d slabs, 2x20KB

    const int lane = threadIdx.x & 31;
    const int wave = threadIdx.x >> 5;
    const int tile = blockIdx.x * 4 + wave;   // 1024 tiles = 4 batches * 256
    const int b    = tile >> 8;
    const int m0   = (tile & 255) * 16;
    const int lo16 = lane & 15;
    const int hi   = lane >> 4;               // half-wave select
    const float alpha = alpha_p[0];

    // ---- A matrix: feat_b [16 x 32] in contiguous-k A layout ----
    // lanes 0-15: K {0..15}; lanes 16-31: K {16..31}; row = lo16
    const _Float16* fbp = featb + ((size_t)b * N_ + m0 + lo16) * CQ;
    v8h a_lo = *(const v8h*)(fbp + hi * 16);
    v8h a_hi = *(const v8h*)(fbp + hi * 16 + 8);
    v16h fbA;
    #pragma unroll
    for (int i = 0; i < 8; ++i) { fbA[i] = a_lo[i]; fbA[i + 8] = a_hi[i]; }

    // ---- constant all-ones B operand for the row-sum WMMA ----
    v16h onesB;
    #pragma unroll
    for (int i = 0; i < 16; ++i) onesB[i] = (_Float16)1.0f;

    // ---- accumulators: 16 x 256 output tile + row-sum tile ----
    v8f acc[16];
    const v8f vzero = {0.f, 0.f, 0.f, 0.f, 0.f, 0.f, 0.f, 0.f};
    #pragma unroll
    for (int ct = 0; ct < 16; ++ct) acc[ct] = vzero;
    v8f accL = vzero;                          // accL[r] = l[m0 + r + 8*hi]

    const _Float16* fcB = featcT + (size_t)b * N_ * CQ;
    const _Float16* fdB = featd  + (size_t)b * C_ * N_;
    _Float16* pl = plds[wave];

#if USE_TDM
    if (wave == 0) {
        tdm_load_slab(fdB, dbuf[0]);           // preload slab 0
        __builtin_amdgcn_s_wait_tensorcnt(0);
    }
    __syncthreads();
#endif

    for (int kt = 0; kt < N_ / 32; ++kt) {
        const int k0  = kt * 32;
        const int cur = kt & 1;

#if USE_TDM
        if (wave == 0 && kt + 1 < N_ / 32)      // async DMA of next slab
            tdm_load_slab(fdB + k0 + 32, dbuf[cur ^ 1]);
#endif

        // ---- logits: S[16 x 32] via two 16x16x32 WMMAs -------------------
        // B layout: lane = column, VGPR j holds K = kbase+2j, kbase+2j+1.
        v16h bc0, bc1;
        __builtin_memcpy(&bc0, fcB + (size_t)(k0 +      lo16) * CQ + hi * 16, 32);
        __builtin_memcpy(&bc1, fcB + (size_t)(k0 + 16 + lo16) * CQ + hi * 16, 32);
        if (kt + 1 < N_ / 32)
            __builtin_prefetch(fcB + (size_t)(k0 + 32 + lo16) * CQ, 0, 1);

        v8f s0 = __builtin_amdgcn_wmma_f32_16x16x32_f16(
            false, fbA, false, bc0, (short)0, vzero, false, false);
        v8f s1 = __builtin_amdgcn_wmma_f32_16x16x32_f16(
            false, fbA, false, bc1, (short)0, vzero, false, false);

        // ---- unnormalized P = exp2(S2): bare v_exp_f32 (inputs bounded) --
        #pragma unroll
        for (int r = 0; r < 8; ++r) {
            int m = r + 8 * hi;
            pl[m * 32 +      lo16] = (_Float16)__builtin_amdgcn_exp2f(s0[r]);
            pl[m * 32 + 16 + lo16] = (_Float16)__builtin_amdgcn_exp2f(s1[r]);
        }
        asm volatile("s_wait_dscnt 0" ::: "memory");

        // ---- reload P in contiguous-k A layout ------------------------------------
        const _Float16* pr = pl + lo16 * 32 + hi * 16;
        v8h pa_lo = *(const v8h*)pr;
        v8h pa_hi = *(const v8h*)(pr + 8);
        v16h pa;
        #pragma unroll
        for (int i = 0; i < 8; ++i) { pa[i] = pa_lo[i]; pa[i + 8] = pa_hi[i]; }

        // ---- row sums l += P @ ones (matrix pipe, no shuffles) -----------
        accL = __builtin_amdgcn_wmma_f32_16x16x32_f16(
            false, pa, false, onesB, (short)0, accL, false, false);

        // ---- PV: O[16 x 256] += P[16 x 32] @ D[32 x 256] (16 WMMAs) ------
        #pragma unroll
        for (int ct = 0; ct < 16; ++ct) {
            v16h bd;
#if USE_TDM
            const _Float16* dp =
                dbuf[cur] + (size_t)(ct * 16 + lo16) * DROW + hi * 16;
            v8h d0 = *(const v8h*)dp;
            v8h d1 = *(const v8h*)(dp + 8);
            #pragma unroll
            for (int i = 0; i < 8; ++i) { bd[i] = d0[i]; bd[i + 8] = d1[i]; }
#else
            __builtin_memcpy(&bd,
                fdB + (size_t)(ct * 16 + lo16) * N_ + k0 + hi * 16, 32);
#endif
            acc[ct] = __builtin_amdgcn_wmma_f32_16x16x32_f16(
                false, pa, false, bd, (short)0, acc[ct], false, false);
        }

#if USE_TDM
        if (wave == 0 && kt + 1 < N_ / 32)
            __builtin_amdgcn_s_wait_tensorcnt(0);   // next slab landed
        __syncthreads();                            // all waves done with cur
#endif
    }

    // ---- epilogue: out = alpha * O / l + x ------------------------------
    float inv[8];
    #pragma unroll
    for (int r = 0; r < 8; ++r) inv[r] = 1.0f / accL[r];

    const float* xb = xin + (size_t)b * C_ * N_;
    float*       ob = out + (size_t)b * C_ * N_;
    #pragma unroll
    for (int ct = 0; ct < 16; ++ct) {
        size_t base = (size_t)(ct * 16 + lo16) * N_ + m0 + 8 * hi;
        float4 x0 = *(const float4*)(xb + base);
        float4 x1 = *(const float4*)(xb + base + 4);
        float4 o0, o1;
        o0.x = fmaf(alpha * inv[0], acc[ct][0], x0.x);
        o0.y = fmaf(alpha * inv[1], acc[ct][1], x0.y);
        o0.z = fmaf(alpha * inv[2], acc[ct][2], x0.z);
        o0.w = fmaf(alpha * inv[3], acc[ct][3], x0.w);
        o1.x = fmaf(alpha * inv[4], acc[ct][4], x1.x);
        o1.y = fmaf(alpha * inv[5], acc[ct][5], x1.y);
        o1.z = fmaf(alpha * inv[6], acc[ct][6], x1.z);
        o1.w = fmaf(alpha * inv[7], acc[ct][7], x1.w);
        *(float4*)(ob + base)     = o0;
        *(float4*)(ob + base + 4) = o1;
    }
}

// ---------------------------------------------------------------------------
extern "C" void kernel_launch(void* const* d_in, const int* in_sizes, int n_in,
                              void* d_out, int out_size, void* d_ws, size_t ws_size,
                              hipStream_t stream) {
    const float* x     = (const float*)d_in[0];
    const float* x1    = (const float*)d_in[1];
    const float* Wb    = (const float*)d_in[2];
    const float* bb    = (const float*)d_in[3];
    const float* Wc    = (const float*)d_in[4];
    const float* bc    = (const float*)d_in[5];
    const float* Wd    = (const float*)d_in[6];
    const float* bd    = (const float*)d_in[7];
    const float* alpha = (const float*)d_in[8];
    float* out = (float*)d_out;

    char* ws = (char*)d_ws;
    _Float16* featb  = (_Float16*)(ws);                 // [B][N][32]  : 1 MB
    _Float16* featcT = (_Float16*)(ws + (1u << 20));    // [B][N][32]  : 1 MB
    _Float16* featd  = (_Float16*)(ws + (2u << 20));    // [B][256][N] : 8 MB

    const float kLog2e = 1.4426950408889634f;  // logits in base-2 domain
    dim3 pb(256), pg(B_ * (N_ / 64));
    proj_kernel<<<pg, pb, 0, stream>>>(x,  Wb, bb, featb,  CQ, 1, kLog2e);
    proj_kernel<<<pg, pb, 0, stream>>>(x1, Wc, bc, featcT, CQ, 1, 1.0f);
    proj_kernel<<<pg, pb, 0, stream>>>(x,  Wd, bd, featd,  C_, 0, 1.0f);

    pam_attn_kernel<<<dim3(256), dim3(128), 0, stream>>>(
        featb, featcT, featd, x, alpha, out);
}
